// dpool_39865886441527
// MI455X (gfx1250) — matrix operand order repaired and checked by reading, not compiled, not run on hardware
//
#include <hip/hip_runtime.h>
#include <hip/hip_bf16.h>
#include <stdint.h>

#define B_   16
#define CIN  256
#define T_   4096
#define C2   512      // conv1 output channels (before GLU)
#define CO   256      // channels after GLU
#define P_   32
#define LP   4110     // (T+2) padded up so (T+2+pad) % 15 == 0
#define EPS1 1e-3f
#define EPSP 1e-5f

typedef _Float16 v16h __attribute__((ext_vector_type(16)));
typedef _Float16 v8h  __attribute__((ext_vector_type(8)));
typedef float    v8f  __attribute__((ext_vector_type(8)));
union V16H { v16h v; v8h h[2]; };

// fast sigmoid: v_exp_f32 + v_rcp_f32 (avoid precise-division div_scale chains)
__device__ __forceinline__ float sigmoidf_(float x) {
  return __builtin_amdgcn_rcpf(1.0f + __expf(-x));
}
__device__ __forceinline__ float siluf_(float x) { return x * sigmoidf_(x); }

// ---------------------------------------------------------------------------
// K0: fold BN1 into conv1 weights -> f16 W' (512x256), and bias' (512)
// ---------------------------------------------------------------------------
__global__ __launch_bounds__(256) void fold_w_kernel(
    const float* __restrict__ conv1_w, const float* __restrict__ g,
    const float* __restrict__ bb, const float* __restrict__ m,
    const float* __restrict__ v, _Float16* __restrict__ Wh,
    float* __restrict__ biasF) {
  int gid = blockIdx.x * 256 + threadIdx.x;          // 512*256 threads
  int c = gid >> 8;
  float scale = g[c] * rsqrtf(v[c] + EPS1);
  Wh[gid] = (_Float16)(conv1_w[gid] * scale);
  if (gid < C2) {
    float sc = g[gid] * rsqrtf(v[gid] + EPS1);
    biasF[gid] = bb[gid] - m[gid] * sc;
  }
}

// ---------------------------------------------------------------------------
// K1: xx[b,t] = sum_c x[b,c,t] * w1x1[c]
// ---------------------------------------------------------------------------
__global__ __launch_bounds__(256) void xx_kernel(
    const float* __restrict__ x, const float* __restrict__ w1,
    float* __restrict__ xx) {
  int gid = blockIdx.x * 256 + threadIdx.x;          // B*T threads
  int b = gid >> 12, t = gid & (T_ - 1);
  const float* xb = x + (size_t)b * CIN * T_ + t;
  float acc = 0.f;
#pragma unroll 4
  for (int c = 0; c < CIN; ++c) acc += xb[(size_t)c * T_] * w1[c];
  xx[gid] = acc;
}

// ---------------------------------------------------------------------------
// K2: p0 conv (Cin=2: [xx, xx^2], K=31, pad=15) + BN + SiLU -> h0 (B,32,T)
// ---------------------------------------------------------------------------
__global__ __launch_bounds__(256) void p0_kernel(
    const float* __restrict__ xx, const float* __restrict__ w,
    const float* __restrict__ pb, const float* __restrict__ g,
    const float* __restrict__ bb, const float* __restrict__ m,
    const float* __restrict__ v, float* __restrict__ h0) {
  int b = blockIdx.y, t0 = blockIdx.x * 128, tid = threadIdx.x;
  __shared__ float sxx[160];                // covers t0-16 .. t0+143
  __shared__ float sw[P_ * 2 * 31];         // 1984 floats
  for (int i = tid; i < 160; i += 256) {
    int t = t0 + i - 16;
    sxx[i] = (t >= 0 && t < T_) ? xx[b * T_ + t] : 0.f;
  }
  for (int i = tid; i < P_ * 2 * 31; i += 256) sw[i] = w[i];
  __syncthreads();
#pragma unroll
  for (int it = 0; it < 16; ++it) {
    int idx = it * 256 + tid;
    int tl = idx & 127, co = idx >> 7;      // co in 0..31
    float acc = pb[co];
    const float* w0 = &sw[(co * 2 + 0) * 31];
    const float* w1 = &sw[(co * 2 + 1) * 31];
#pragma unroll
    for (int k = 0; k < 31; ++k) {
      float h = sxx[tl + k + 1];            // t + k - 15
      acc += h * w0[k] + (h * h) * w1[k];
    }
    float scale = g[co] * rsqrtf(v[co] + EPSP);
    float z = (acc - m[co]) * scale + bb[co];
    h0[((size_t)b * P_ + co) * T_ + t0 + tl] = siluf_(z);
  }
}

// ---------------------------------------------------------------------------
// K3: p1 conv (32->32, K=15, pad=7) + BN + SiLU -> h1 (B,32,T)
// ---------------------------------------------------------------------------
__global__ __launch_bounds__(256) void p1_kernel(
    const float* __restrict__ h0, const float* __restrict__ w,
    const float* __restrict__ pb, const float* __restrict__ g,
    const float* __restrict__ bb, const float* __restrict__ m,
    const float* __restrict__ v, float* __restrict__ h1) {
  int b = blockIdx.y, t0 = blockIdx.x * 128, tid = threadIdx.x;
  __shared__ float sh[P_][144];             // covers t0-8 .. t0+135
  for (int i = tid; i < P_ * 144; i += 256) {
    int ci = i / 144, j = i - ci * 144;
    int t = t0 + j - 8;
    sh[ci][j] = (t >= 0 && t < T_) ? h0[((size_t)b * P_ + ci) * T_ + t] : 0.f;
  }
  __syncthreads();
#pragma unroll
  for (int it = 0; it < 16; ++it) {
    int idx = it * 256 + tid;
    int tl = idx & 127, co = idx >> 7;
    float acc = pb[co];
    for (int ci = 0; ci < P_; ++ci) {
      const float* r = &sh[ci][tl + 1];     // t + k - 7
      const float* wr = &w[((size_t)co * P_ + ci) * 15];
#pragma unroll
      for (int k = 0; k < 15; ++k) acc += r[k] * wr[k];
    }
    float scale = g[co] * rsqrtf(v[co] + EPSP);
    float z = (acc - m[co]) * scale + bb[co];
    h1[((size_t)b * P_ + co) * T_ + t0 + tl] = siluf_(z);
  }
}

// ---------------------------------------------------------------------------
// K4: p2 conv (32->2, K=15, pad=7) -> weights=sig(j0), movesRaw=sig(j1)*nm
// ---------------------------------------------------------------------------
__global__ __launch_bounds__(256) void p2_kernel(
    const float* __restrict__ h1, const float* __restrict__ w,
    const float* __restrict__ pb, const float* __restrict__ nm,
    float* __restrict__ wts, float* __restrict__ mvs) {
  int b = blockIdx.y, t0 = blockIdx.x * 256, tid = threadIdx.x;
  __shared__ float sh[P_][272];             // covers t0-8 .. t0+263
  __shared__ float sw[2 * P_ * 15];
  for (int i = tid; i < P_ * 272; i += 256) {
    int ci = i / 272, j = i - ci * 272;
    int t = t0 + j - 8;
    sh[ci][j] = (t >= 0 && t < T_) ? h1[((size_t)b * P_ + ci) * T_ + t] : 0.f;
  }
  for (int i = tid; i < 2 * P_ * 15; i += 256) sw[i] = w[i];
  __syncthreads();
  int tl = tid;
  float a0 = pb[0], a1 = pb[1];
  for (int ci = 0; ci < P_; ++ci) {
    const float* r = &sh[ci][tl + 1];
    const float* w0 = &sw[(0 * P_ + ci) * 15];
    const float* w1 = &sw[(1 * P_ + ci) * 15];
#pragma unroll
    for (int k = 0; k < 15; ++k) { a0 += r[k] * w0[k]; a1 += r[k] * w1[k]; }
  }
  int gid = b * T_ + t0 + tl;
  wts[gid] = sigmoidf_(a0);
  mvs[gid] = sigmoidf_(a1) * nm[0];
}

// ---------------------------------------------------------------------------
// K5: per-row renorm + inclusive scan -> s1 = (1-frac)*wt, s2 = frac*wt, fi
// (keeps IEEE division for poses: frac feeds floorf -> bin selection)
// ---------------------------------------------------------------------------
__global__ __launch_bounds__(1024) void scan_kernel(
    const float* __restrict__ mvs, const float* __restrict__ wts,
    float* __restrict__ s1, float* __restrict__ s2, int* __restrict__ fi) {
  int b = blockIdx.x, tid = threadIdx.x;    // 1024 threads, 4 elems each
  const float* mrow = mvs + (size_t)b * T_;
  float4 mv = *(const float4*)(mrow + tid * 4);
  float c0 = mv.x, c1 = c0 + mv.y, c2 = c1 + mv.z, c3 = c2 + mv.w;
  float s = c3;
  // wave32 inclusive scan of per-thread sums
  float xinc = s;
  int lane = tid & 31, wave = tid >> 5;
#pragma unroll
  for (int d = 1; d < 32; d <<= 1) {
    float y = __shfl_up(xinc, d, 32);
    if (lane >= d) xinc += y;
  }
  __shared__ float wsum[32];
  if (lane == 31) wsum[wave] = xinc;
  __syncthreads();
  if (wave == 0) {
    float y = wsum[lane];
#pragma unroll
    for (int d = 1; d < 32; d <<= 1) {
      float z = __shfl_up(y, d, 32);
      if (lane >= d) y += z;
    }
    wsum[lane] = y;
  }
  __syncthreads();
  float wprefix = (wave == 0) ? 0.f : wsum[wave - 1];
  float total = wsum[31];
  float excl = wprefix + xinc - s;          // exclusive prefix before this thread
  float renorm = fmaxf(total * (1.0f / T_), 1.0f);
  float inv = 1.0f / renorm;                // precise: feeds floorf
  float pos[4] = {excl + c0, excl + c1, excl + c2, excl + c3};
  float4 wv = *(const float4*)(wts + (size_t)b * T_ + tid * 4);
  float wl[4] = {wv.x, wv.y, wv.z, wv.w};
#pragma unroll
  for (int j = 0; j < 4; ++j) {
    float p = pos[j] * inv;
    float fl = floorf(p);
    float frac = p - fl;
    int idx = b * T_ + tid * 4 + j;
    s1[idx] = (1.0f - frac) * wl[j];
    s2[idx] = frac * wl[j];
    fi[idx] = (int)fl;
  }
}

// ---------------------------------------------------------------------------
// K6: zero the output (also covers the length-padding region)
// ---------------------------------------------------------------------------
__global__ __launch_bounds__(256) void zero_kernel(float* __restrict__ o, int n) {
  int gid = blockIdx.x * 256 + threadIdx.x;
  if (gid < n) o[gid] = 0.f;
}

// ---------------------------------------------------------------------------
// K7: WMMA GEMM (f16 in, f32 acc) + GLU + linear-interp scatter-add.
// One block = ALL 256 GLU channels (512 GEMM rows) x 64 t, so x is staged to
// LDS exactly once device-wide (x traffic = 64 MB, the roofline minimum).
// Block: 256 threads = 8 waves; wave w: channels [w*32, w*32+32) as 2 m-tiles
// (a rows + g rows share the B fragment). 16 WMMAs / k-step / wave.
// ---------------------------------------------------------------------------
#define LSTR 40   // LDS halves per n-row (32 k + pad), 80B -> 16B aligned rows
__global__ __launch_bounds__(256) void gemm_glu_scatter_kernel(
    const float* __restrict__ x, const _Float16* __restrict__ Wh,
    const float* __restrict__ biasF, const float* __restrict__ s1,
    const float* __restrict__ s2, const int* __restrict__ fi,
    float* __restrict__ out) {
  __shared__ __align__(16) _Float16 ldsX[64 * LSTR];
  const int tid  = threadIdx.x;
  const int lane = tid & 31, wave = tid >> 5;     // wave 0..7
  const int lo = lane & 15, hi = lane >> 4;
  const int n_block = blockIdx.x * 64;
  const int b  = n_block >> 12;                   // 4096 % 64 == 0
  const int t0 = n_block & (T_ - 1);
  const int c_wave = wave * 32;                   // 32 channels per wave
  const float* xb = x + (size_t)b * CIN * T_ + t0;

  const v8f vzero = {0.f, 0.f, 0.f, 0.f, 0.f, 0.f, 0.f, 0.f};
  v8f accA[2][4], accG[2][4];
#pragma unroll
  for (int mt = 0; mt < 2; ++mt)
#pragma unroll
    for (int nt = 0; nt < 4; ++nt) { accA[mt][nt] = vzero; accG[mt][nt] = vzero; }

  const int k_l = tid >> 3;                       // 0..31  (k row within tile)
  const int colbase = (tid & 7) * 8;              // 0..56  (n column group)

  for (int kk = 0; kk < 8; ++kk) {
    const int k0 = kk * 32;
    // stage X tile (32 k x 64 n) fp32 -> f16 into LDS as [n][k]
    const float* gp = xb + (size_t)(k0 + k_l) * T_ + colbase;
    float4 v0 = ((const float4*)gp)[0];
    float4 v1 = ((const float4*)gp)[1];
    float tmp[8] = {v0.x, v0.y, v0.z, v0.w, v1.x, v1.y, v1.z, v1.w};
#pragma unroll
    for (int j = 0; j < 8; ++j)
      ldsX[(colbase + j) * LSTR + k_l] = (_Float16)tmp[j];
    __syncthreads();

    // A fragments (ISA 16-bit A 16x32): lanes<16 k={0..7,16..23}, lanes>=16 +8
    V16H aA[2], aG[2];
#pragma unroll
    for (int mt = 0; mt < 2; ++mt) {
      const _Float16* wa =
          Wh + (size_t)(c_wave + mt * 16 + lo) * CIN + k0 + hi * 8;
      aA[mt].h[0] = *(const v8h*)(wa);
      aA[mt].h[1] = *(const v8h*)(wa + 16);
      const _Float16* wg = wa + (size_t)CO * CIN;  // GLU gate rows c+256
      aG[mt].h[0] = *(const v8h*)(wg);
      aG[mt].h[1] = *(const v8h*)(wg + 16);
    }

#pragma unroll
    for (int nt = 0; nt < 4; ++nt) {
      // B fragment (ISA 16-bit B 32x16): n = lane&15, k = {0..7,16..23} + hi*8
      const _Float16* bp = &ldsX[(nt * 16 + lo) * LSTR + hi * 8];
      V16H bf;
      bf.h[0] = *(const v8h*)(bp);
      bf.h[1] = *(const v8h*)(bp + 16);
#pragma unroll
      for (int mt = 0; mt < 2; ++mt) {
        accA[mt][nt] = __builtin_amdgcn_wmma_f32_16x16x32_f16(
            false, aA[mt].v, false, bf.v, (short)0, accA[mt][nt], false, false);
        accG[mt][nt] = __builtin_amdgcn_wmma_f32_16x16x32_f16(
            false, aG[mt].v, false, bf.v, (short)0, accG[mt][nt], false, false);
      }
    }
    __syncthreads();
  }

  // Epilogue: GLU + scatter. D layout: lane->t (N=lo), VGPR r -> c (+8 if hi).
  const size_t outB = (size_t)b * CO * LP;
#pragma unroll
  for (int nt = 0; nt < 4; ++nt) {
    const int t = t0 + nt * 16 + lo;
    const int bt = b * T_ + t;
    const float sv1 = s1[bt], sv2 = s2[bt];
    const int p = fi[bt];
#pragma unroll
    for (int mt = 0; mt < 2; ++mt) {
#pragma unroll
      for (int r = 0; r < 8; ++r) {
        const int c = c_wave + mt * 16 + hi * 8 + r;
        float av = accA[mt][nt][r] + biasF[c];
        float gv = accG[mt][nt][r] + biasF[c + CO];
        float feat = av * sigmoidf_(gv);
        float* po = out + outB + (size_t)c * LP + p;
        __hip_atomic_fetch_add(po,     feat * sv1, __ATOMIC_RELAXED,
                               __HIP_MEMORY_SCOPE_AGENT);
        __hip_atomic_fetch_add(po + 1, feat * sv2, __ATOMIC_RELAXED,
                               __HIP_MEMORY_SCOPE_AGENT);
      }
    }
  }
}

// ---------------------------------------------------------------------------
extern "C" void kernel_launch(void* const* d_in, const int* in_sizes, int n_in,
                              void* d_out, int out_size, void* d_ws,
                              size_t ws_size, hipStream_t stream) {
  const float* x        = (const float*)d_in[0];
  const float* conv1_w  = (const float*)d_in[1];
  const float* bn1_g    = (const float*)d_in[2];
  const float* bn1_b    = (const float*)d_in[3];
  const float* bn1_m    = (const float*)d_in[4];
  const float* bn1_v    = (const float*)d_in[5];
  const float* w1x1     = (const float*)d_in[6];
  const float* p0_w     = (const float*)d_in[7];
  const float* p0_b     = (const float*)d_in[8];
  const float* p0bn_g   = (const float*)d_in[9];
  const float* p0bn_b   = (const float*)d_in[10];
  const float* p0bn_m   = (const float*)d_in[11];
  const float* p0bn_v   = (const float*)d_in[12];
  const float* p1_w     = (const float*)d_in[13];
  const float* p1_b     = (const float*)d_in[14];
  const float* p1bn_g   = (const float*)d_in[15];
  const float* p1bn_b   = (const float*)d_in[16];
  const float* p1bn_m   = (const float*)d_in[17];
  const float* p1bn_v   = (const float*)d_in[18];
  const float* p2_w     = (const float*)d_in[19];
  const float* p2_b     = (const float*)d_in[20];
  const float* nm       = (const float*)d_in[21];
  float* out = (float*)d_out;

  // carve workspace (all 256B aligned)
  uintptr_t base = (uintptr_t)d_ws;
  auto carve = [&](size_t bytes) {
    base = (base + 255) & ~(uintptr_t)255;
    uintptr_t r = base;
    base += bytes;
    return (void*)r;
  };
  _Float16* Wh  = (_Float16*)carve((size_t)C2 * CIN * 2);
  float* biasF  = (float*)carve(C2 * 4);
  float* xx     = (float*)carve((size_t)B_ * T_ * 4);
  float* h0     = (float*)carve((size_t)B_ * P_ * T_ * 4);
  float* h1     = (float*)carve((size_t)B_ * P_ * T_ * 4);
  float* wts    = (float*)carve((size_t)B_ * T_ * 4);
  float* mvs    = (float*)carve((size_t)B_ * T_ * 4);
  float* s1     = (float*)carve((size_t)B_ * T_ * 4);
  float* s2     = (float*)carve((size_t)B_ * T_ * 4);
  int*   fi     = (int*)carve((size_t)B_ * T_ * 4);

  fold_w_kernel<<<(C2 * CIN) / 256, 256, 0, stream>>>(
      conv1_w, bn1_g, bn1_b, bn1_m, bn1_v, Wh, biasF);
  xx_kernel<<<(B_ * T_) / 256, 256, 0, stream>>>(x, w1x1, xx);
  p0_kernel<<<dim3(T_ / 128, B_), 256, 0, stream>>>(
      xx, p0_w, p0_b, p0bn_g, p0bn_b, p0bn_m, p0bn_v, h0);
  p1_kernel<<<dim3(T_ / 128, B_), 256, 0, stream>>>(
      h0, p1_w, p1_b, p1bn_g, p1bn_b, p1bn_m, p1bn_v, h1);
  p2_kernel<<<dim3(T_ / 256, B_), 256, 0, stream>>>(h1, p2_w, p2_b, nm, wts, mvs);
  scan_kernel<<<B_, 1024, 0, stream>>>(mvs, wts, s1, s2, fi);
  zero_kernel<<<(out_size + 255) / 256, 256, 0, stream>>>(out, out_size);
  gemm_glu_scatter_kernel<<<(B_ * T_) / 64, 256, 0, stream>>>(
      x, Wh, biasF, s1, s2, fi, out);
}
